// MyEmbeddingModel_80015240725023
// MI455X (gfx1250) — compile-verified, matrix operand order checked by reading, not run on hardware
//
#include <hip/hip_runtime.h>

#define NN 50000
#define EE 600000
#define DD 128

typedef __attribute__((ext_vector_type(16))) _Float16 v16h;
typedef __attribute__((ext_vector_type(8)))  float    v8f;

// ---------------- zero travel + degree ----------------
__global__ void init_kernel(float* __restrict__ travel, int* __restrict__ deg) {
  int i = blockIdx.x * blockDim.x + threadIdx.x;
  if (i < NN) { travel[i] = 0.0f; deg[i] = 0; }
}

// ---------------- travel[src] += w5*dist ; deg[src]++ ----------------
__global__ void edge_stats_kernel(const int* __restrict__ src,
                                  const float* __restrict__ dist,
                                  const float* __restrict__ w5p,
                                  float* __restrict__ travel,
                                  int* __restrict__ deg) {
  int e = blockIdx.x * blockDim.x + threadIdx.x;
  if (e < EE) {
    int s = src[e];
    atomicAdd(&travel[s], w5p[0] * dist[e]);
    atomicAdd(&deg[s], 1);
  }
}

// ---------------- exclusive scan of degrees -> row_ptr, cursor (1 block) ----------------
__global__ void scan_kernel(const int* __restrict__ deg,
                            int* __restrict__ row_ptr,
                            int* __restrict__ cursor) {
  __shared__ int s[1024];
  const int t = threadIdx.x;
  const int chunk = (NN + 1023) / 1024;     // 49
  const int b0 = t * chunk;
  int sum = 0;
  for (int c = 0; c < chunk; ++c) { int i = b0 + c; if (i < NN) sum += deg[i]; }
  s[t] = sum;
  __syncthreads();
  for (int off = 1; off < 1024; off <<= 1) {   // Hillis-Steele inclusive scan
    int v = (t >= off) ? s[t - off] : 0;
    __syncthreads();
    s[t] += v;
    __syncthreads();
  }
  int run = s[t] - sum;                        // exclusive prefix for this chunk
  for (int c = 0; c < chunk; ++c) {
    int i = b0 + c;
    if (i < NN) { row_ptr[i] = run; cursor[i] = run; run += deg[i]; }
  }
  if (t == 1023) row_ptr[NN] = s[1023];
}

// ---------------- scatter edges into CSR col array ----------------
__global__ void fill_kernel(const int* __restrict__ src,
                            const int* __restrict__ dst,
                            int* __restrict__ cursor,
                            int* __restrict__ col) {
  int e = blockIdx.x * blockDim.x + threadIdx.x;
  if (e < EE) {
    int p = atomicAdd(&cursor[src[e]], 1);
    col[p] = dst[e];
  }
}

// ---------------- base = w1*x + [relu(travel),1] @ [[w4*1],[w2*g]] via WMMA ----------------
// One wave per 16x16 output tile; 3125 row-tiles x 8 col-tiles = 25000 tiles.
__global__ void base_wmma_kernel(const float* __restrict__ x,
                                 const float* __restrict__ g,
                                 const float* __restrict__ travel,
                                 const float* __restrict__ w1p,
                                 const float* __restrict__ w2p,
                                 const float* __restrict__ w4p,
                                 float* __restrict__ base) {
  const int lane = threadIdx.x & 31;
  const int tile = blockIdx.x * (blockDim.x >> 5) + (threadIdx.x >> 5);
  const int tm = tile >> 3;              // 0..3124
  const int tn = tile & 7;               // 0..7
  const int row0 = tm << 4, col0 = tn << 4;
  const float w1 = w1p[0];

  // A (16x32 f16): K=0 -> relu(travel[m]), K=1 -> 1.0, rest 0.
  // Lanes 0-15 hold K=0..7,16..23 (so a[0]=K0, a[1]=K1); lanes 16-31 hold only K>=8 -> zero.
  // B (32x16 f16): K=0 row -> w4, K=1 row -> w2*g[n], rest 0.
  // Lanes 0-15 hold K=0..7,16..23 (b[0]=K0, b[1]=K1); lanes 16-31 hold only K>=8 -> zero.
  v16h a = {};
  v16h b = {};
  if (lane < 16) {
    float t = fmaxf(travel[row0 + lane], 0.0f);
    a[0] = (_Float16)t;
    a[1] = (_Float16)1.0f;
    b[0] = (_Float16)w4p[0];
    b[1] = (_Float16)(w2p[0] * g[col0 + lane]);
  }

  // C/D 16x16 f32 layout: lanes 0-15 -> N=lane, VGPR v -> M=v; lanes 16-31 -> M=8+v.
  const int cc = col0 + (lane & 15);
  const int rb = row0 + ((lane < 16) ? 0 : 8);
  v8f c;
#pragma unroll
  for (int v = 0; v < 8; ++v) c[v] = w1 * x[(size_t)(rb + v) * DD + cc];

  c = __builtin_amdgcn_wmma_f32_16x16x32_f16(
      /*neg_a=*/false, a, /*neg_b=*/false, b,
      /*c_mod=*/(short)0, c, /*reuse_a=*/false, /*reuse_b=*/false);

#pragma unroll
  for (int v = 0; v < 8; ++v) base[(size_t)(rb + v) * DD + cc] = c[v];
}

// ---------------- one propagation round: uout = base + w3 * sum_{nbr} uin[nbr] ----------------
// One wave32 per node; lane holds 4 consecutive floats of the 128-dim embedding.
__global__ void prop_kernel(const float* __restrict__ uin,
                            const float* __restrict__ basev,
                            const int* __restrict__ row_ptr,
                            const int* __restrict__ col,
                            const float* __restrict__ w3p,
                            float* __restrict__ uout,
                            const int do_relu) {
  const int lane = threadIdx.x & 31;
  const int node = (blockIdx.x * blockDim.x + threadIdx.x) >> 5;  // exact: 6250*8 = 50000
  const int d4 = lane << 2;

  const int beg = row_ptr[node];
  const int end = row_ptr[node + 1];

  float4 acc = make_float4(0.f, 0.f, 0.f, 0.f);
  int j = (beg < end) ? col[beg] : 0;
  for (int k = beg; k < end; ++k) {
    const int jn = (k + 1 < end) ? col[k + 1] : 0;
    __builtin_prefetch(uin + (size_t)jn * DD + d4, 0, 1);   // global_prefetch_b8
    const float4 uv = *(const float4*)(uin + (size_t)j * DD + d4);
    acc.x += uv.x; acc.y += uv.y; acc.z += uv.z; acc.w += uv.w;
    j = jn;
  }

  const float w3 = w3p[0];
  const size_t off = (size_t)node * DD + d4;
  const float4 bv = *(const float4*)(basev + off);
  float4 r;
  r.x = fmaf(w3, acc.x, bv.x);
  r.y = fmaf(w3, acc.y, bv.y);
  r.z = fmaf(w3, acc.z, bv.z);
  r.w = fmaf(w3, acc.w, bv.w);
  if (do_relu) {
    r.x = fmaxf(r.x, 0.f); r.y = fmaxf(r.y, 0.f);
    r.z = fmaxf(r.z, 0.f); r.w = fmaxf(r.w, 0.f);
  }
  *(float4*)(uout + off) = r;
}

extern "C" void kernel_launch(void* const* d_in, const int* in_sizes, int n_in,
                              void* d_out, int out_size, void* d_ws, size_t ws_size,
                              hipStream_t stream) {
  const float* x    = (const float*)d_in[0];
  const float* g    = (const float*)d_in[1];
  const float* dist = (const float*)d_in[2];
  const float* w1   = (const float*)d_in[3];
  const float* w2   = (const float*)d_in[4];
  const float* w3   = (const float*)d_in[5];
  const float* w4   = (const float*)d_in[6];
  const float* w5   = (const float*)d_in[7];
  const int*   src  = (const int*)d_in[8];
  const int*   dst  = (const int*)d_in[9];
  float* out = (float*)d_out;
  (void)in_sizes; (void)n_in; (void)out_size; (void)ws_size;

  char* ws = (char*)d_ws;
  size_t off = 0;
  auto alloc = [&](size_t bytes) -> void* {
    void* p = (void*)(ws + off);
    off = (off + bytes + 255) & ~(size_t)255;
    return p;
  };
  float* travel  = (float*)alloc((size_t)NN * 4);
  int*   deg     = (int*)  alloc((size_t)NN * 4);
  int*   row_ptr = (int*)  alloc((size_t)(NN + 1) * 4);
  int*   cursor  = (int*)  alloc((size_t)NN * 4);
  int*   col     = (int*)  alloc((size_t)EE * 4);
  float* base    = (float*)alloc((size_t)NN * DD * 4);
  float* uA      = (float*)alloc((size_t)NN * DD * 4);
  float* uB      = (float*)alloc((size_t)NN * DD * 4);

  init_kernel      <<<(NN + 255) / 256, 256, 0, stream>>>(travel, deg);
  edge_stats_kernel<<<(EE + 255) / 256, 256, 0, stream>>>(src, dist, w5, travel, deg);
  scan_kernel      <<<1, 1024, 0, stream>>>(deg, row_ptr, cursor);
  fill_kernel      <<<(EE + 255) / 256, 256, 0, stream>>>(src, dst, cursor, col);

  // 25000 tiles, 4 waves/block -> 6250 blocks exactly (full EXEC at WMMA).
  base_wmma_kernel<<<6250, 128, 0, stream>>>(x, g, travel, w1, w2, w4, base);

  // P = 5 rounds, ping-pong; last round fuses ReLU and writes d_out.
  prop_kernel<<<6250, 256, 0, stream>>>(x,  base, row_ptr, col, w3, uA, 0);
  prop_kernel<<<6250, 256, 0, stream>>>(uA, base, row_ptr, col, w3, uB, 0);
  prop_kernel<<<6250, 256, 0, stream>>>(uB, base, row_ptr, col, w3, uA, 0);
  prop_kernel<<<6250, 256, 0, stream>>>(uA, base, row_ptr, col, w3, uB, 0);
  prop_kernel<<<6250, 256, 0, stream>>>(uB, base, row_ptr, col, w3, out, 1);
}
